// PointerGeneratorNetwork_54606214201415
// MI455X (gfx1250) — matrix operand order repaired and checked by reading, not compiled, not run on hardware
//
#include <hip/hip_runtime.h>
#include <hip/hip_bf16.h>
#include <math.h>

// ---- problem constants -----------------------------------------------------
#define B_     8
#define TE_    400
#define TD_    100
#define E_     128
#define H_     256
#define H2_    512      // 2H
#define G4H_   1024     // 4H
#define A_     612
#define V_     50000
#define MAXOOV_ 50
#define VEXT_  50050

// vocab-softmax tiling
#define NSLICE 25
#define CPS    2000       // columns per slice (50000/25)
#define TPS    125        // 16-wide tiles per slice
#define KPAD256 264       // padded K stride (f16) for 16x256 A tile (528B, 16B-mult)
#define KPAD512 520       // padded K stride (f16) for 16x512 A tile (1040B, 16B-mult)

typedef __attribute__((ext_vector_type(16))) _Float16 v16h;
typedef __attribute__((ext_vector_type(8)))  _Float16 h8;
typedef __attribute__((ext_vector_type(8)))  float    v8f;

__device__ __forceinline__ float sigf(float x) { return 1.0f / (1.0f + expf(-x)); }

__device__ __forceinline__ v8f vzero8() {
  v8f z = {0.f, 0.f, 0.f, 0.f, 0.f, 0.f, 0.f, 0.f};
  return z;
}

// Wave computes a 16x16 f32 tile of A(16xK f16 in LDS, row stride kpad) x
// B(f16, TRANSPOSED: Bt[col][k], col stride ldbt) with v_wmma_f32_16x16x32_f16.
// A-frag (16-bit A 16x32): lane<16 -> M=lane, K-half 0; lane>=16 -> K-half 8.
// B-frag (32x16): N = lane&15; same K order as A (lane>=16 -> K-half 8).
// Per lane per WMMA: 2x ds_load_b128 (A, hoisted across tiles) + 2x global_load_b128 (B).
__device__ __forceinline__ v8f wmma_tile_t(const _Float16* __restrict__ aLds, int kpad,
                                           const _Float16* __restrict__ Bt, int ldbt,
                                           int col, int kTotal, int lane, v8f acc) {
  const int m  = lane & 15;
  const int hi = lane >> 4;
  const _Float16* ap = aLds + m * kpad + (hi ? 8 : 0);
  const _Float16* bp = Bt + (size_t)col * (size_t)ldbt + (hi ? 8 : 0);
#pragma unroll
  for (int k0 = 0; k0 < kTotal; k0 += 32) {
    h8 a0 = *(const h8*)(ap + k0);
    h8 a1 = *(const h8*)(ap + k0 + 16);
    h8 b0 = *(const h8*)(bp + k0);
    h8 b1 = *(const h8*)(bp + k0 + 16);
    v16h af = __builtin_shufflevector(a0, a1, 0, 1, 2, 3, 4, 5, 6, 7,
                                      8, 9, 10, 11, 12, 13, 14, 15);
    v16h bf = __builtin_shufflevector(b0, b1, 0, 1, 2, 3, 4, 5, 6, 7,
                                      8, 9, 10, 11, 12, 13, 14, 15);
    acc = __builtin_amdgcn_wmma_f32_16x16x32_f16(false, af, false, bf,
                                                 (short)0, acc, false, false);
  }
  return acc;
}

// ---- K0: transpose + f32->f16 convert: W[K][N] -> Wt[N][K] ------------------
__global__ void k_transpose_h(const float* __restrict__ W, _Float16* __restrict__ Wt,
                              int K, int N) {
  __shared__ _Float16 tl[64][66];     // 66-half pad -> conflict-free strides
  const int nbc = (N + 63) >> 6;
  const int bc = blockIdx.x % nbc;
  const int bk = blockIdx.x / nbc;
  const int c0 = bc * 64, k0 = bk * 64;
  const int tid = threadIdx.x;        // 256
  for (int i = tid; i < 64 * 64; i += 256) {
    const int r = i >> 6, c = i & 63;         // coalesced over c
    const int col = c0 + c;
    const float v = (col < N) ? W[(size_t)(k0 + r) * N + col] : 0.0f;
    tl[r][c] = (_Float16)v;
  }
  __syncthreads();
  for (int i = tid; i < 64 * 64; i += 256) {
    const int c = i >> 6, r = i & 63;         // coalesced over r (contig k)
    const int col = c0 + c;
    if (col < N) Wt[(size_t)col * K + (k0 + r)] = tl[r][c];
  }
}

// ---- K1: embedding gather --------------------------------------------------
__global__ void k_gather(const int* __restrict__ etok, const int* __restrict__ dtok,
                         const float* __restrict__ emb,
                         float* __restrict__ enc_emb, float* __restrict__ dec_emb) {
  int idx = blockIdx.x * blockDim.x + threadIdx.x;
  const int nEnc = B_ * TE_ * E_;
  const int nDec = TD_ * B_ * E_;
  if (idx < nEnc) {
    int e = idx % E_;
    int t = (idx / E_) % TE_;
    int b = idx / (E_ * TE_);
    enc_emb[idx] = emb[(size_t)etok[b * TE_ + t] * E_ + e];
  } else if (idx < nEnc + nDec) {
    int j = idx - nEnc;
    int e = j % E_;
    int b = (j / E_) % B_;
    int t = j / (E_ * B_);
    dec_emb[j] = emb[(size_t)dtok[b * TD_ + t] * E_ + e];
  }
}

// ---- K2: bidirectional encoder LSTM (16 blocks = dir x batch, 256 thr) -----
__global__ void k_enc_lstm(const float* __restrict__ enc_emb,
                           const float* __restrict__ Wih_f, const float* __restrict__ Whh_f,
                           const float* __restrict__ b_f,
                           const float* __restrict__ Wih_b, const float* __restrict__ Whh_b,
                           const float* __restrict__ b_b,
                           float* __restrict__ enc_out, float* __restrict__ hT) {
  __shared__ float xsh[E_];
  __shared__ float hsh[H_];
  const int dir = blockIdx.x >> 3;
  const int b   = blockIdx.x & 7;
  const int u   = threadIdx.x;          // 0..255
  const float* Wih = dir ? Wih_b : Wih_f;
  const float* Whh = dir ? Whh_b : Whh_f;
  const float* bb  = dir ? b_b  : b_f;

  float creg = 0.0f;
  hsh[u] = 0.0f;
  const float bi = bb[u], bf2 = bb[H_ + u], bg = bb[2 * H_ + u], bo = bb[3 * H_ + u];
  __syncthreads();

  for (int s = 0; s < TE_; ++s) {
    const int t = dir ? (TE_ - 1 - s) : s;
    if (u < E_) xsh[u] = enc_emb[((size_t)b * TE_ + t) * E_ + u];
    __syncthreads();
    float gi = bi, gf = bf2, gg = bg, go = bo;
    for (int e = 0; e < E_; ++e) {
      const float xv = xsh[e];
      const float* w = Wih + (size_t)e * G4H_;
      gi += xv * w[u]; gf += xv * w[H_ + u]; gg += xv * w[2 * H_ + u]; go += xv * w[3 * H_ + u];
    }
    for (int j = 0; j < H_; ++j) {
      const float hv = hsh[j];
      const float* w = Whh + (size_t)j * G4H_;
      gi += hv * w[u]; gf += hv * w[H_ + u]; gg += hv * w[2 * H_ + u]; go += hv * w[3 * H_ + u];
    }
    creg = sigf(gf) * creg + sigf(gi) * tanhf(gg);
    const float hn = sigf(go) * tanhf(creg);
    __syncthreads();
    hsh[u] = hn;
    enc_out[((size_t)b * TE_ + t) * H2_ + dir * H_ + u] = hn;
    __syncthreads();
  }
  // hT layout: [dir][h/c][b][H]
  hT[(size_t)dir * 2 * B_ * H_ + (size_t)b * H_ + u]           = hsh[u];
  hT[(size_t)dir * 2 * B_ * H_ + B_ * H_ + (size_t)b * H_ + u] = creg;
}

// ---- K2b: h0/c0 reduction --------------------------------------------------
__global__ void k_h0c0(const float* __restrict__ hT, const float* __restrict__ Wred,
                       const float* __restrict__ bred,
                       float* __restrict__ h0, float* __restrict__ c0) {
  int idx = blockIdx.x * blockDim.x + threadIdx.x;
  if (idx >= 2 * B_ * H_) return;
  const int which = idx / (B_ * H_);   // 0 -> h0, 1 -> c0
  const int r = idx % (B_ * H_);
  const int b = r / H_, u = r % H_;
  const float* hf = hT + (which ? B_ * H_ : 0) + (size_t)b * H_;
  const float* hb = hT + 2 * B_ * H_ + (which ? B_ * H_ : 0) + (size_t)b * H_;
  float s = bred[u];
  for (int j = 0; j < H_; ++j) s += hf[j] * Wred[(size_t)j * H_ + u];
  for (int j = 0; j < H_; ++j) s += hb[j] * Wred[(size_t)(H_ + j) * H_ + u];
  s = fmaxf(s, 0.0f);
  (which ? c0 : h0)[b * H_ + u] = s;
}

// ---- K3: enc_feat = enc_out @ Wenc_feat + b (WMMA, transposed f16 B) -------
__global__ void __launch_bounds__(256, 1)
k_encfeat(const float* __restrict__ enc_out, const _Float16* __restrict__ WefT,
          const float* __restrict__ bef, float* __restrict__ enc_feat) {
  __shared__ _Float16 aLds[16 * KPAD512];
  const int rg  = blockIdx.x;           // 200 groups of 16 rows over B*TE=3200
  const int tid = threadIdx.x;          // 256
  for (int idx = tid; idx < 16 * H2_; idx += 256) {
    int i = idx / H2_, k = idx % H2_;
    aLds[i * KPAD512 + k] = (_Float16)enc_out[((size_t)rg * 16 + i) * H2_ + k];
  }
  __syncthreads();
  const int w = tid >> 5, lane = tid & 31, m = lane & 15, hi = lane >> 4;
  for (int ct = w; ct < 39; ct += 8) {
    const int col0 = ct * 16;
    const int col  = col0 + m;
    const int colc = (col < A_) ? col : (A_ - 1);   // clamp: no predicated loads
    v8f acc = vzero8();
    acc = wmma_tile_t(aLds, KPAD512, WefT, H2_, colc, H2_, lane, acc);
    if (col < A_) {
      const float bb = bef[col];
#pragma unroll
      for (int vg = 0; vg < 8; ++vg) {
        const int row = rg * 16 + vg + hi * 8;
        enc_feat[(size_t)row * A_ + col] = acc[vg] + bb;
      }
    }
  }
}

// ---- K4: decoder with attention (8 blocks = batch rows, 512 thr) -----------
__global__ void k_decoder(const float* __restrict__ dec_emb, const float* __restrict__ enc_out,
                          const float* __restrict__ enc_feat, const float* __restrict__ mask,
                          const float* __restrict__ vvec,
                          const float* __restrict__ Wx, const float* __restrict__ bx,
                          const float* __restrict__ Wih_d, const float* __restrict__ Whh_d,
                          const float* __restrict__ b_d,
                          const float* __restrict__ Wdf, const float* __restrict__ bdf,
                          const float* __restrict__ Wpg, const float* __restrict__ bpg,
                          const float* __restrict__ Wout, const float* __restrict__ bout,
                          const float* __restrict__ h0, const float* __restrict__ c0,
                          float* __restrict__ outs2, float* __restrict__ pgens,
                          float* __restrict__ attns) {
  __shared__ float hsh[H_], csh[H_], ctx[H2_], xsh[E_], embt[E_];
  __shared__ float gsh[G4H_], dfeat[A_], esh[TE_], red[512], vsh[A_];
  const int b   = blockIdx.x;
  const int tid = threadIdx.x;          // 512
  if (tid < H_)  { hsh[tid] = h0[b * H_ + tid]; csh[tid] = c0[b * H_ + tid]; }
  if (tid < H2_) ctx[tid] = 0.0f;
  for (int a = tid; a < A_; a += 512) vsh[a] = vvec[a];
  __syncthreads();

  for (int t = 0; t < TD_; ++t) {
    if (tid < E_) embt[tid] = dec_emb[((size_t)t * B_ + b) * E_ + tid];
    __syncthreads();
    // x = concat(emb, ctx) @ Wx + bx
    if (tid < E_) {
      float s = bx[tid];
      for (int e = 0; e < E_; ++e)  s += embt[e] * Wx[(size_t)e * E_ + tid];
      for (int j = 0; j < H2_; ++j) s += ctx[j] * Wx[(size_t)(E_ + j) * E_ + tid];
      xsh[tid] = s;
    }
    __syncthreads();
    // gates
    for (int cc = tid; cc < G4H_; cc += 512) {
      float s = b_d[cc];
      for (int e = 0; e < E_; ++e) s += xsh[e] * Wih_d[(size_t)e * G4H_ + cc];
      for (int j = 0; j < H_; ++j) s += hsh[j] * Whh_d[(size_t)j * G4H_ + cc];
      gsh[cc] = s;
    }
    __syncthreads();
    if (tid < H_) {
      const float gi = gsh[tid], gf = gsh[H_ + tid], gg = gsh[2 * H_ + tid], go = gsh[3 * H_ + tid];
      const float cn = sigf(gf) * csh[tid] + sigf(gi) * tanhf(gg);
      csh[tid] = cn;
      hsh[tid] = sigf(go) * tanhf(cn);
    }
    __syncthreads();
    // dec_feat = concat(h,c) @ Wdf[t] + bdf[t]
    for (int a = tid; a < A_; a += 512) {
      float s = bdf[(size_t)t * A_ + a];
      const float* wt = Wdf + (size_t)t * H2_ * A_;
      for (int j = 0; j < H_; ++j) s += hsh[j] * wt[(size_t)j * A_ + a];
      for (int j = 0; j < H_; ++j) s += csh[j] * wt[(size_t)(H_ + j) * A_ + a];
      dfeat[a] = s;
    }
    __syncthreads();
    // attention energies
    for (int tt = tid; tt < TE_; tt += 512) {
      const float* ef = enc_feat + ((size_t)b * TE_ + tt) * A_;
      float s = 0.0f;
      for (int a = 0; a < A_; ++a) s += vsh[a] * tanhf(ef[a] + dfeat[a]);
      esh[tt] = s;
    }
    __syncthreads();
    // masked softmax + renorm
    float pm = -1e30f;
    for (int tt = tid; tt < TE_; tt += 512) pm = fmaxf(pm, esh[tt]);
    red[tid] = pm;
    __syncthreads();
    for (int off = 256; off > 0; off >>= 1) {
      if (tid < off) red[tid] = fmaxf(red[tid], red[tid + off]);
      __syncthreads();
    }
    const float mx = red[0];
    __syncthreads();
    float ps = 0.0f;
    for (int tt = tid; tt < TE_; tt += 512) {
      const float wv = expf(esh[tt] - mx) * mask[b * TE_ + tt];
      esh[tt] = wv;
      ps += wv;
    }
    red[tid] = ps;
    __syncthreads();
    for (int off = 256; off > 0; off >>= 1) {
      if (tid < off) red[tid] += red[tid + off];
      __syncthreads();
    }
    const float inv = 1.0f / red[0];
    __syncthreads();
    for (int tt = tid; tt < TE_; tt += 512) {
      const float av = esh[tt] * inv;
      esh[tt] = av;
      attns[((size_t)t * B_ + b) * TE_ + tt] = av;
    }
    __syncthreads();
    // context
    if (tid < H2_) {
      float s = 0.0f;
      for (int tt = 0; tt < TE_; ++tt) s += esh[tt] * enc_out[((size_t)b * TE_ + tt) * H2_ + tid];
      ctx[tid] = s;
    }
    __syncthreads();
    // p_gen
    float pp = 0.0f;
    for (int idx = tid; idx < H2_ + 2 * H_ + E_; idx += 512) {
      float z = (idx < H2_)          ? ctx[idx]
              : (idx < H2_ + H_)     ? hsh[idx - H2_]
              : (idx < H2_ + 2 * H_) ? csh[idx - H2_ - H_]
                                     : xsh[idx - H2_ - 2 * H_];
      pp += z * Wpg[idx];
    }
    red[tid] = pp;
    __syncthreads();
    for (int off = 256; off > 0; off >>= 1) {
      if (tid < off) red[tid] += red[tid + off];
      __syncthreads();
    }
    if (tid == 0) pgens[t * B_ + b] = sigf(red[0] + bpg[0]);
    __syncthreads();
    // out = concat(h, ctx) @ Wout + bout
    if (tid < H_) {
      float s = bout[tid];
      for (int j = 0; j < H_; ++j)  s += hsh[j] * Wout[(size_t)j * H_ + tid];
      for (int j = 0; j < H2_; ++j) s += ctx[j] * Wout[(size_t)(H_ + j) * H_ + tid];
      outs2[((size_t)t * B_ + b) * H_ + tid] = s;
    }
    __syncthreads();
  }
}

// ---- K5: vocab logits pass 1 — online (max,sum) per row via WMMA -----------
__global__ void __launch_bounds__(256, 1)
k_vstats(const float* __restrict__ outs2, const _Float16* __restrict__ WvT,
         const float* __restrict__ bv, float* __restrict__ partials) {
  __shared__ _Float16 aLds[16 * KPAD256];
  __shared__ float sM[8][16], sS[8][16];
  const int rg  = blockIdx.x / NSLICE;
  const int sl  = blockIdx.x % NSLICE;
  const int tid = threadIdx.x;          // 256
  for (int idx = tid; idx < 16 * H_; idx += 256) {
    int i = idx / H_, k = idx % H_;
    aLds[i * KPAD256 + k] = (_Float16)outs2[((size_t)rg * 16 + i) * H_ + k];
  }
  __syncthreads();
  const int w = tid >> 5, lane = tid & 31, m = lane & 15, hi = lane >> 4;
  float mrun[8], srun[8];
#pragma unroll
  for (int vg = 0; vg < 8; ++vg) { mrun[vg] = -1e30f; srun[vg] = 0.0f; }

  for (int ct = w; ct < TPS; ct += 8) {
    const int col = sl * CPS + ct * 16 + m;
    if (ct + 8 < TPS)  // prefetch next B tile for this wave
      __builtin_prefetch(WvT + (size_t)(col + 128) * H_, 0, 1);
    v8f acc = vzero8();
    acc = wmma_tile_t(aLds, KPAD256, WvT, H_, col, H_, lane, acc);
    const float bvv = bv[col];
#pragma unroll
    for (int vg = 0; vg < 8; ++vg) {
      const float val = acc[vg] + bvv;
      const float nm  = fmaxf(mrun[vg], val);
      srun[vg] = srun[vg] * expf(mrun[vg] - nm) + expf(val - nm);
      mrun[vg] = nm;
    }
  }
  // butterfly reduce across the 16 lanes sharing each row
#pragma unroll
  for (int off = 1; off < 16; off <<= 1) {
#pragma unroll
    for (int vg = 0; vg < 8; ++vg) {
      const float om = __shfl_xor(mrun[vg], off, 16);
      const float os = __shfl_xor(srun[vg], off, 16);
      const float nm = fmaxf(mrun[vg], om);
      srun[vg] = srun[vg] * expf(mrun[vg] - nm) + os * expf(om - nm);
      mrun[vg] = nm;
    }
  }
  if (m == 0) {
#pragma unroll
    for (int vg = 0; vg < 8; ++vg) { sM[w][vg + hi * 8] = mrun[vg]; sS[w][vg + hi * 8] = srun[vg]; }
  }
  __syncthreads();
  if (tid < 16) {
    float mm = -1e30f, ss = 0.0f;
    for (int ww = 0; ww < 8; ++ww) {
      const float om = sM[ww][tid], os = sS[ww][tid];
      const float nm = fmaxf(mm, om);
      ss = ss * expf(mm - nm) + os * expf(om - nm);
      mm = nm;
    }
    partials[(size_t)blockIdx.x * 32 + tid * 2]     = mm;
    partials[(size_t)blockIdx.x * 32 + tid * 2 + 1] = ss;
  }
}

// ---- K6: merge per-slice stats --------------------------------------------
__global__ void k_vreduce(const float* __restrict__ partials, float* __restrict__ rowstats) {
  const int row = blockIdx.x * blockDim.x + threadIdx.x;
  if (row >= TD_ * B_) return;
  const int rg = row >> 4, r = row & 15;
  float mm = -1e30f, ss = 0.0f;
  for (int sl = 0; sl < NSLICE; ++sl) {
    const float* p = partials + (size_t)(rg * NSLICE + sl) * 32 + r * 2;
    const float om = p[0], os = p[1];
    const float nm = fmaxf(mm, om);
    ss = ss * expf(mm - nm) + os * expf(om - nm);
    mm = nm;
  }
  rowstats[row * 2]     = mm;
  rowstats[row * 2 + 1] = ss;
}

// ---- K7: vocab pass 2 — recompute tiles, write pgen*softmax ----------------
__global__ void __launch_bounds__(256, 1)
k_vwrite(const float* __restrict__ outs2, const _Float16* __restrict__ WvT,
         const float* __restrict__ bv, const float* __restrict__ rowstats,
         const float* __restrict__ pgens, float* __restrict__ finalp) {
  __shared__ _Float16 aLds[16 * KPAD256];
  const int rg  = blockIdx.x / NSLICE;
  const int sl  = blockIdx.x % NSLICE;
  const int tid = threadIdx.x;          // 256
  for (int idx = tid; idx < 16 * H_; idx += 256) {
    int i = idx / H_, k = idx % H_;
    aLds[i * KPAD256 + k] = (_Float16)outs2[((size_t)rg * 16 + i) * H_ + k];
  }
  __syncthreads();
  const int w = tid >> 5, lane = tid & 31, m = lane & 15, hi = lane >> 4;
  float mr[8], sr[8], pgr[8];
  int rowr[8];
#pragma unroll
  for (int vg = 0; vg < 8; ++vg) {
    const int row = rg * 16 + vg + hi * 8;
    rowr[vg] = row;
    mr[vg]   = rowstats[row * 2];
    sr[vg]   = 1.0f / rowstats[row * 2 + 1];
    pgr[vg]  = pgens[row];
  }
  for (int ct = w; ct < TPS; ct += 8) {
    const int col = sl * CPS + ct * 16 + m;
    if (ct + 8 < TPS)
      __builtin_prefetch(WvT + (size_t)(col + 128) * H_, 0, 1);
    v8f acc = vzero8();
    acc = wmma_tile_t(aLds, KPAD256, WvT, H_, col, H_, lane, acc);
    const float bvv = bv[col];
#pragma unroll
    for (int vg = 0; vg < 8; ++vg) {
      const float val = acc[vg] + bvv;
      finalp[(size_t)rowr[vg] * VEXT_ + col] = pgr[vg] * expf(val - mr[vg]) * sr[vg];
    }
  }
}

// ---- K7b: zero OOV tail of output ------------------------------------------
__global__ void k_zero_oov(float* __restrict__ finalp) {
  const int idx = blockIdx.x * blockDim.x + threadIdx.x;
  if (idx >= TD_ * B_ * MAXOOV_) return;
  const int row = idx / MAXOOV_, j = idx % MAXOOV_;
  finalp[(size_t)row * VEXT_ + V_ + j] = 0.0f;
}

// ---- K8: copy-distribution scatter ------------------------------------------
__global__ void k_scatter(const float* __restrict__ attns, const float* __restrict__ pgens,
                          const int* __restrict__ ext, float* __restrict__ finalp) {
  const int t = blockIdx.x / B_, b = blockIdx.x % B_;
  const float wgt = 1.0f - pgens[t * B_ + b];
  for (int j = threadIdx.x; j < TE_; j += blockDim.x) {
    const float a = attns[((size_t)t * B_ + b) * TE_ + j] * wgt;
    const int idx = ext[b * TE_ + j];
    atomicAdd(&finalp[((size_t)t * B_ + b) * VEXT_ + idx], a);
  }
}

// ---- launch -----------------------------------------------------------------
extern "C" void kernel_launch(void* const* d_in, const int* in_sizes, int n_in,
                              void* d_out, int out_size, void* d_ws, size_t ws_size,
                              hipStream_t stream) {
  (void)in_sizes; (void)n_in; (void)out_size; (void)ws_size;
  const int*   etok  = (const int*)d_in[0];
  const int*   dtok  = (const int*)d_in[1];
  const int*   ext   = (const int*)d_in[2];
  const float* mask  = (const float*)d_in[3];
  const float* emb   = (const float*)d_in[4];
  const float* Wih_f = (const float*)d_in[5];
  const float* Whh_f = (const float*)d_in[6];
  const float* b_f   = (const float*)d_in[7];
  const float* Wih_b = (const float*)d_in[8];
  const float* Whh_b = (const float*)d_in[9];
  const float* b_b   = (const float*)d_in[10];
  const float* Wred  = (const float*)d_in[11];
  const float* bred  = (const float*)d_in[12];
  const float* Wef   = (const float*)d_in[13];
  const float* bef   = (const float*)d_in[14];
  const float* vvec  = (const float*)d_in[15];
  const float* Wx    = (const float*)d_in[16];
  const float* bx    = (const float*)d_in[17];
  const float* Wih_d = (const float*)d_in[18];
  const float* Whh_d = (const float*)d_in[19];
  const float* b_d   = (const float*)d_in[20];
  const float* Wdf   = (const float*)d_in[21];
  const float* bdf   = (const float*)d_in[22];
  const float* Wpg   = (const float*)d_in[23];
  const float* bpg   = (const float*)d_in[24];
  const float* Wout  = (const float*)d_in[25];
  const float* bout  = (const float*)d_in[26];
  const float* Wv    = (const float*)d_in[27];
  const float* bv    = (const float*)d_in[28];
  float* finalp = (float*)d_out;

  float* ws = (float*)d_ws;
  size_t o = 0;
  float* enc_emb  = ws + o; o += (size_t)B_ * TE_ * E_;     // 409600
  float* dec_emb  = ws + o; o += (size_t)TD_ * B_ * E_;     // 102400
  float* enc_out  = ws + o; o += (size_t)B_ * TE_ * H2_;    // 1638400
  float* hT       = ws + o; o += (size_t)4 * B_ * H_;       // 8192
  float* h0       = ws + o; o += (size_t)B_ * H_;           // 2048
  float* c0       = ws + o; o += (size_t)B_ * H_;           // 2048
  float* enc_feat = ws + o; o += (size_t)B_ * TE_ * A_;     // 1958400
  float* outs2    = ws + o; o += (size_t)TD_ * B_ * H_;     // 204800
  float* pgens    = ws + o; o += (size_t)TD_ * B_;          // 800
  float* attns    = ws + o; o += (size_t)TD_ * B_ * TE_;    // 320000
  float* partials = ws + o; o += (size_t)50 * NSLICE * 32;  // 40000
  float* rowstats = ws + o; o += (size_t)TD_ * B_ * 2;      // 1600
  o = (o + 3) & ~(size_t)3;                                 // 16B align for f16 regions
  _Float16* WvT  = (_Float16*)(ws + o); o += ((size_t)V_ * H_) / 2;   // 6.4M floats
  _Float16* WefT = (_Float16*)(ws + o); o += ((size_t)A_ * H2_ + 1) / 2;

  // weight transposes/conversions (depend only on inputs)
  k_transpose_h<<<((V_ + 63) / 64) * (H_ / 64), 256, 0, stream>>>(Wv, WvT, H_, V_);
  k_transpose_h<<<((A_ + 63) / 64) * (H2_ / 64), 256, 0, stream>>>(Wef, WefT, H2_, A_);

  const int nGather = B_ * TE_ * E_ + TD_ * B_ * E_;
  k_gather<<<(nGather + 255) / 256, 256, 0, stream>>>(etok, dtok, emb, enc_emb, dec_emb);
  k_enc_lstm<<<16, 256, 0, stream>>>(enc_emb, Wih_f, Whh_f, b_f, Wih_b, Whh_b, b_b, enc_out, hT);
  k_h0c0<<<16, 256, 0, stream>>>(hT, Wred, bred, h0, c0);
  k_encfeat<<<200, 256, 0, stream>>>(enc_out, WefT, bef, enc_feat);
  k_decoder<<<8, 512, 0, stream>>>(dec_emb, enc_out, enc_feat, mask, vvec,
                                   Wx, bx, Wih_d, Whh_d, b_d, Wdf, bdf,
                                   Wpg, bpg, Wout, bout, h0, c0,
                                   outs2, pgens, attns);
  k_vstats<<<50 * NSLICE, 256, 0, stream>>>(outs2, WvT, bv, partials);
  k_vreduce<<<(TD_ * B_ + 255) / 256, 256, 0, stream>>>(partials, rowstats);
  k_vwrite<<<50 * NSLICE, 256, 0, stream>>>(outs2, WvT, bv, rowstats, pgens, finalp);
  k_zero_oov<<<(TD_ * B_ * MAXOOV_ + 255) / 256, 256, 0, stream>>>(finalp);
  k_scatter<<<TD_ * B_, 128, 0, stream>>>(attns, pgens, ext, finalp);
}
